// MultiHeadAttention_34016140984995
// MI455X (gfx1250) — matrix operand order repaired and checked by reading, not compiled, not run on hardware
//
#include <hip/hip_runtime.h>
#include <hip/hip_bf16.h>

// MI455X / gfx1250 wave32 WMMA multi-head attention.
// B=4, S=2048, E=1024, H=16, DH=64. Compute-bound (~137 GFLOP vs ~100MB HBM),
// so all GEMMs (QKV proj, Q*K^T, P*V, out proj) run through
// v_wmma_f32_16x16x32_f16 with f32 accumulation.

#define B_  4
#define S_  2048
#define E_  1024
#define H_  16
#define DH_ 64

typedef _Float16 half16v __attribute__((ext_vector_type(16)));
typedef _Float16 half8v  __attribute__((ext_vector_type(8)));
typedef _Float16 half4v  __attribute__((ext_vector_type(4)));
typedef float    float8v __attribute__((ext_vector_type(8)));

static __device__ __forceinline__ float fast_exp2(float x) {
  return __builtin_amdgcn_exp2f(x);  // v_exp_f32 (TRANS pipe)
}

static __device__ __forceinline__ half16v cat8(half8v lo, half8v hi) {
  return __builtin_shufflevector(lo, hi, 0,1,2,3,4,5,6,7,8,9,10,11,12,13,14,15);
}

static __device__ __forceinline__ float8v wmma_f16(half16v a, half16v b, float8v c) {
  // (neg_a, A, neg_b, B, c_mod, C, reuse_a, reuse_b)
  return __builtin_amdgcn_wmma_f32_16x16x32_f16(false, a, false, b, (short)0, c,
                                                false, false);
}

// A fragment (16x32 f16, MxK) from row-major matrix; base points at (row0,k0),
// leading dim `ld` in halves. Layout used here: lane m=lane%16 holds row m,
// K = (lane/16)*16 + {0..15} (contiguous, matches B fragments).
static __device__ __forceinline__ half16v load_a(const _Float16* base, int ld, int lane) {
  const int m  = lane & 15;
  const int kb = (lane >> 4) << 4;
  const _Float16* p = base + (size_t)m * ld + kb;
  half8v lo = *(const half8v*)(p);
  half8v hi = *(const half8v*)(p + 8);
  return cat8(lo, hi);
}

// B fragment (32x16, KxN) where B[k][n] = M[n0+n][k0+k] and M is row-major with
// leading dim `ld` (halves). Layout used here: lane n holds column n%16,
// K = (lane/16)*16 + e  -> 16 contiguous halves of M's row n.
static __device__ __forceinline__ half16v load_bT(const _Float16* Mbase, int ld, int lane) {
  const int n  = lane & 15;
  const int kb = (lane >> 4) << 4;
  return *(const half16v*)(Mbase + (size_t)n * ld + kb);
}

// ---------------------------------------------------------------------------
// Kernel 0: f32 -> f16 conversion (x, Wo). n must be a multiple of 4.
// ---------------------------------------------------------------------------
__global__ void f32_to_f16_kernel(const float* __restrict__ src,
                                  _Float16* __restrict__ dst, int n) {
  const int n4 = n >> 2;
  const int stride = gridDim.x * blockDim.x;
  for (int i = blockIdx.x * blockDim.x + threadIdx.x; i < n4; i += stride) {
    float4 v = ((const float4*)src)[i];
    half4v o;
    o[0] = (_Float16)v.x; o[1] = (_Float16)v.y;
    o[2] = (_Float16)v.z; o[3] = (_Float16)v.w;
    ((half4v*)dst)[i] = o;
  }
}

// ---------------------------------------------------------------------------
// Kernel 1: pre-pack Wq/Wk/Wv [H,E,DH] f32 into per-lane WMMA B-fragment
// order: [h][nt(4)][kt(32)][lane(32)][e(16)] f16, so the GEMM's B-fragment
// load is one contiguous 32B load per lane.
// ---------------------------------------------------------------------------
__global__ void pack_w_kernel(const float* __restrict__ Wq,
                              const float* __restrict__ Wk,
                              const float* __restrict__ Wv,
                              _Float16* __restrict__ wqp,
                              _Float16* __restrict__ wkp,
                              _Float16* __restrict__ wvp) {
  const int tid  = blockIdx.x * blockDim.x + threadIdx.x; // 3<<20 threads
  const int e    = tid & 15;
  const int lane = (tid >> 4) & 31;
  const int kt   = (tid >> 9) & 31;
  const int nt   = (tid >> 14) & 3;
  const int h    = (tid >> 16) & 15;
  const int t    = tid >> 20;
  if (t > 2) return;
  const int kd  = kt * 32 + ((lane >> 4) << 4) + e;
  const int col = nt * 16 + (lane & 15);
  const float* W = (t == 0) ? Wq : (t == 1) ? Wk : Wv;
  _Float16* outp = (t == 0) ? wqp : (t == 1) ? wkp : wvp;
  const float val = W[((size_t)h * E_ + kd) * DH_ + col];
  outp[(((((size_t)h * 4 + nt) * 32 + kt) * 32) + lane) * 16 + e] = (_Float16)val;
}

// ---------------------------------------------------------------------------
// Kernel 2: QKV projection. One block = 4 waves = 128x64 output tile for one
// (b,h,t). Q,K stored row-major f16 [BH][S][DH]; V written directly in packed
// B-fragment order (via wave-private LDS bounce) for the later P*V GEMM.
// ---------------------------------------------------------------------------
__global__ __launch_bounds__(128) void qkv_kernel(
    const _Float16* __restrict__ xh,
    const _Float16* __restrict__ wqp, const _Float16* __restrict__ wkp,
    const _Float16* __restrict__ wvp,
    const float* __restrict__ bq, const float* __restrict__ bk,
    const float* __restrict__ bv,
    _Float16* __restrict__ qout, _Float16* __restrict__ kout,
    _Float16* __restrict__ vpk) {
  __shared__ _Float16 vbuf[4][32][64];

  const int lane = threadIdx.x & 31;
  const int wave = threadIdx.x >> 5;
  const int mt   = blockIdx.x;   // 0..S/128-1
  const int bh   = blockIdx.y;   // b*H + h
  const int t    = blockIdx.z;   // 0=q 1=k 2=v
  const int b    = bh >> 4;
  const int h    = bh & 15;
  const int row0 = mt * 128 + wave * 32;

  const _Float16* wp   = (t == 0) ? wqp : (t == 1) ? wkp : wvp;
  const float*    bias = ((t == 0) ? bq : (t == 1) ? bk : bv) + h * DH_;
  const _Float16* wbase = wp + (size_t)h * (4 * 32 * 512);

  float8v c[2][4];
  #pragma unroll
  for (int s = 0; s < 2; ++s)
    #pragma unroll
    for (int n = 0; n < 4; ++n)
      #pragma unroll
      for (int v = 0; v < 8; ++v) c[s][n][v] = 0.0f;

  for (int kt = 0; kt < E_ / 32; ++kt) {
    half16v a0 = load_a(xh + ((size_t)(b * S_ + row0 +  0)) * E_ + kt * 32, E_, lane);
    half16v a1 = load_a(xh + ((size_t)(b * S_ + row0 + 16)) * E_ + kt * 32, E_, lane);
    #pragma unroll
    for (int nt = 0; nt < 4; ++nt) {
      half16v bf = *(const half16v*)(wbase + ((size_t)(nt * 32 + kt)) * 512 + lane * 16);
      c[0][nt] = wmma_f16(a0, bf, c[0][nt]);
      c[1][nt] = wmma_f16(a1, bf, c[1][nt]);
    }
  }

  const int hi8 = (lane >> 4) << 3;
  if (t < 2) {
    _Float16* outp = (t == 0) ? qout : kout;
    #pragma unroll
    for (int s = 0; s < 2; ++s)
      #pragma unroll
      for (int nt = 0; nt < 4; ++nt) {
        const int col = nt * 16 + (lane & 15);
        const float bcol = bias[col];
        #pragma unroll
        for (int v = 0; v < 8; ++v) {
          const int srow = row0 + s * 16 + hi8 + v;
          outp[((size_t)bh * S_ + srow) * DH_ + col] = (_Float16)(c[s][nt][v] + bcol);
        }
      }
  } else {
    // V: C-layout -> packed B-fragment layout via LDS (wave-private slab).
    #pragma unroll
    for (int s = 0; s < 2; ++s)
      #pragma unroll
      for (int nt = 0; nt < 4; ++nt) {
        const int col = nt * 16 + (lane & 15);
        const float bcol = bias[col];
        #pragma unroll
        for (int v = 0; v < 8; ++v)
          vbuf[wave][s * 16 + hi8 + v][col] = (_Float16)(c[s][nt][v] + bcol);
      }
    __syncthreads();
    const int kt2 = row0 >> 5;  // 32-key tile index within S
    #pragma unroll
    for (int nt = 0; nt < 4; ++nt) {
      half16v o;
      #pragma unroll
      for (int e = 0; e < 16; ++e)
        o[e] = vbuf[wave][((lane >> 4) << 4) + e][nt * 16 + (lane & 15)];
      *(half16v*)(vpk + (((size_t)bh * (S_ / 32) + kt2) * 4 + nt) * 512 + lane * 16) = o;
    }
  }
}

// ---------------------------------------------------------------------------
// Kernel 3: flash attention per (b,h). Wave = 32 query rows; loop over 32-key
// tiles; online softmax (exp2 domain); P bounced C-layout -> A-layout in LDS.
// Q fragments are hoisted out of the key loop (loop-invariant), V fragments
// loaded once per key tile.
// ---------------------------------------------------------------------------
__global__ __launch_bounds__(128) void attn_kernel(
    const _Float16* __restrict__ q, const _Float16* __restrict__ kmat,
    const _Float16* __restrict__ vpk, _Float16* __restrict__ concat) {
  __shared__ _Float16 pbuf[4][32][32];

  const int lane = threadIdx.x & 31;
  const int wave = threadIdx.x >> 5;
  const int bh   = blockIdx.y;
  const int b    = bh >> 4;
  const int h    = bh & 15;
  const int row0 = blockIdx.x * 128 + wave * 32;
  const int hi8  = (lane >> 4) << 3;

  const float SC = 0.125f * 1.44269504088896340736f;  // 1/sqrt(64) * log2(e)

  float8v acc[2][4];
  float mrow[2][8], lrow[2][8];
  #pragma unroll
  for (int s = 0; s < 2; ++s) {
    #pragma unroll
    for (int n = 0; n < 4; ++n)
      #pragma unroll
      for (int v = 0; v < 8; ++v) acc[s][n][v] = 0.0f;
    #pragma unroll
    for (int v = 0; v < 8; ++v) { mrow[s][v] = -3.0e38f; lrow[s][v] = 0.0f; }
  }

  const _Float16* qbase = q    + ((size_t)bh * S_ + row0) * DH_;
  const _Float16* kbase = kmat + (size_t)bh * S_ * DH_;
  const _Float16* vbase = vpk  + (size_t)bh * (S_ / 32) * 4 * 512;

  // Q tile (32x64) is invariant over the key loop: load its 4 A-fragments once.
  half16v qa[2][2];
  #pragma unroll
  for (int s = 0; s < 2; ++s)
    #pragma unroll
    for (int ks = 0; ks < 2; ++ks)
      qa[s][ks] = load_a(qbase + (size_t)(s * 16) * DH_ + ks * 32, DH_, lane);

  for (int kt2 = 0; kt2 < S_ / 32; ++kt2) {
    // ---- scores tile S = Q(32x64) * K(32 keys x 64)^T ----
    float8v sc[2][2];
    #pragma unroll
    for (int s = 0; s < 2; ++s)
      #pragma unroll
      for (int n = 0; n < 2; ++n)
        #pragma unroll
        for (int v = 0; v < 8; ++v) sc[s][n][v] = 0.0f;

    #pragma unroll
    for (int ks = 0; ks < 2; ++ks) {
      #pragma unroll
      for (int nk = 0; nk < 2; ++nk) {
        half16v bf = load_bT(kbase + ((size_t)(kt2 * 32 + nk * 16)) * DH_ + ks * 32,
                             DH_, lane);
        sc[0][nk] = wmma_f16(qa[0][ks], bf, sc[0][nk]);
        sc[1][nk] = wmma_f16(qa[1][ks], bf, sc[1][nk]);
      }
    }
    #pragma unroll
    for (int s = 0; s < 2; ++s)
      #pragma unroll
      for (int nk = 0; nk < 2; ++nk)
        #pragma unroll
        for (int v = 0; v < 8; ++v) sc[s][nk][v] *= SC;

    // ---- online softmax update (row stats reduced over 16-lane halves) ----
    #pragma unroll
    for (int s = 0; s < 2; ++s)
      #pragma unroll
      for (int v = 0; v < 8; ++v) {
        float r = fmaxf(sc[s][0][v], sc[s][1][v]);
        r = fmaxf(r, __shfl_xor(r, 1, 32));
        r = fmaxf(r, __shfl_xor(r, 2, 32));
        r = fmaxf(r, __shfl_xor(r, 4, 32));
        r = fmaxf(r, __shfl_xor(r, 8, 32));
        const float mnew = fmaxf(mrow[s][v], r);
        const float corr = fast_exp2(mrow[s][v] - mnew);
        mrow[s][v] = mnew;
        lrow[s][v] *= corr;
        #pragma unroll
        for (int nt = 0; nt < 4; ++nt) acc[s][nt][v] *= corr;
        float ps = 0.0f;
        #pragma unroll
        for (int nk = 0; nk < 2; ++nk) {
          const float p = fast_exp2(sc[s][nk][v] - mnew);
          sc[s][nk][v] = p;
          ps += p;
        }
        ps += __shfl_xor(ps, 1, 32);
        ps += __shfl_xor(ps, 2, 32);
        ps += __shfl_xor(ps, 4, 32);
        ps += __shfl_xor(ps, 8, 32);
        lrow[s][v] += ps;
      }

    // ---- P: C-layout -> LDS -> A-fragment layout ----
    #pragma unroll
    for (int s = 0; s < 2; ++s)
      #pragma unroll
      for (int nk = 0; nk < 2; ++nk)
        #pragma unroll
        for (int v = 0; v < 8; ++v)
          pbuf[wave][s * 16 + hi8 + v][nk * 16 + (lane & 15)] = (_Float16)sc[s][nk][v];

    // ---- acc += P(32x32) * V(32x64), V pre-packed as B fragments ----
    half16v vb[4];
    #pragma unroll
    for (int nt = 0; nt < 4; ++nt)
      vb[nt] = *(const half16v*)(vbase + (((size_t)kt2 * 4) + nt) * 512 + lane * 16);

    #pragma unroll
    for (int s = 0; s < 2; ++s) {
      const _Float16* pr = &pbuf[wave][s * 16 + (lane & 15)][(lane >> 4) << 4];
      half8v lo = *(const half8v*)pr;
      half8v hi = *(const half8v*)(pr + 8);
      half16v pa = cat8(lo, hi);
      #pragma unroll
      for (int nt = 0; nt < 4; ++nt)
        acc[s][nt] = wmma_f16(pa, vb[nt], acc[s][nt]);
    }
  }

  // ---- normalize and store f16 into concat [B*S, E] (head h block) ----
  #pragma unroll
  for (int s = 0; s < 2; ++s)
    #pragma unroll
    for (int v = 0; v < 8; ++v) {
      const float inv = 1.0f / lrow[s][v];
      const int srow = row0 + s * 16 + hi8 + v;
      #pragma unroll
      for (int nt = 0; nt < 4; ++nt) {
        const int col = h * DH_ + nt * 16 + (lane & 15);
        concat[((size_t)(b * S_ + srow)) * E_ + col] = (_Float16)(acc[s][nt][v] * inv);
      }
    }
}

// ---------------------------------------------------------------------------
// Kernel 4: output projection out = concat @ Wo^T + bo (f32 out).
// Wo row-major [E,E]: B[k][n] = Wo[n][k] is contiguous -> direct load_bT.
// ---------------------------------------------------------------------------
__global__ __launch_bounds__(128) void oproj_kernel(
    const _Float16* __restrict__ concat, const _Float16* __restrict__ woh,
    const float* __restrict__ bo, float* __restrict__ out) {
  const int lane = threadIdx.x & 31;
  const int wave = threadIdx.x >> 5;
  const int col0 = blockIdx.x * 64;
  const int row0 = blockIdx.y * 128 + wave * 32;

  float8v c[2][4];
  #pragma unroll
  for (int s = 0; s < 2; ++s)
    #pragma unroll
    for (int n = 0; n < 4; ++n)
      #pragma unroll
      for (int v = 0; v < 8; ++v) c[s][n][v] = 0.0f;

  for (int kt = 0; kt < E_ / 32; ++kt) {
    half16v a0 = load_a(concat + ((size_t)row0) * E_ + kt * 32, E_, lane);
    half16v a1 = load_a(concat + ((size_t)(row0 + 16)) * E_ + kt * 32, E_, lane);
    #pragma unroll
    for (int nt = 0; nt < 4; ++nt) {
      half16v bf = load_bT(woh + ((size_t)(col0 + nt * 16)) * E_ + kt * 32, E_, lane);
      c[0][nt] = wmma_f16(a0, bf, c[0][nt]);
      c[1][nt] = wmma_f16(a1, bf, c[1][nt]);
    }
  }

  const int hi8 = (lane >> 4) << 3;
  #pragma unroll
  for (int s = 0; s < 2; ++s)
    #pragma unroll
    for (int nt = 0; nt < 4; ++nt) {
      const int col = col0 + nt * 16 + (lane & 15);
      const float bc = bo[col];
      #pragma unroll
      for (int v = 0; v < 8; ++v) {
        const int r = row0 + s * 16 + hi8 + v;
        out[(size_t)r * E_ + col] = c[s][nt][v] + bc;
      }
    }
}

// ---------------------------------------------------------------------------
extern "C" void kernel_launch(void* const* d_in, const int* in_sizes, int n_in,
                              void* d_out, int out_size, void* d_ws, size_t ws_size,
                              hipStream_t stream) {
  (void)in_sizes; (void)n_in; (void)out_size; (void)ws_size;
  const float* x  = (const float*)d_in[0];
  const float* Wq = (const float*)d_in[1];
  const float* bq = (const float*)d_in[2];
  const float* Wk = (const float*)d_in[3];
  const float* bk = (const float*)d_in[4];
  const float* Wv = (const float*)d_in[5];
  const float* bv = (const float*)d_in[6];
  const float* Wo = (const float*)d_in[7];
  const float* bo = (const float*)d_in[8];
  float* out = (float*)d_out;

  char* ws = (char*)d_ws;
  size_t off = 0;
  auto alloc = [&](size_t bytes) -> char* {
    char* p = ws + off;
    off += (bytes + 255) & ~(size_t)255;
    return p;
  };
  _Float16* xh   = (_Float16*)alloc((size_t)B_ * S_ * E_ * 2);        // 16 MB
  _Float16* wqp  = (_Float16*)alloc((size_t)H_ * E_ * DH_ * 2);       //  2 MB
  _Float16* wkp  = (_Float16*)alloc((size_t)H_ * E_ * DH_ * 2);       //  2 MB
  _Float16* wvp  = (_Float16*)alloc((size_t)H_ * E_ * DH_ * 2);       //  2 MB
  _Float16* woh  = (_Float16*)alloc((size_t)E_ * E_ * 2);             //  2 MB
  _Float16* qbuf = (_Float16*)alloc((size_t)B_ * H_ * S_ * DH_ * 2);  // 16 MB
  _Float16* kbuf = (_Float16*)alloc((size_t)B_ * H_ * S_ * DH_ * 2);  // 16 MB
  _Float16* vpk  = (_Float16*)alloc((size_t)B_ * H_ * S_ * DH_ * 2);  // 16 MB
  _Float16* cbuf = (_Float16*)alloc((size_t)B_ * S_ * E_ * 2);        // 16 MB

  const int n_x = B_ * S_ * E_;
  f32_to_f16_kernel<<<1024, 256, 0, stream>>>(x, xh, n_x);
  const int n_wo = E_ * E_;
  f32_to_f16_kernel<<<256, 256, 0, stream>>>(Wo, woh, n_wo);

  const int np = 3 * H_ * 4 * 32 * 32 * 16;  // 3<<20
  pack_w_kernel<<<np / 256, 256, 0, stream>>>(Wq, Wk, Wv, wqp, wkp, wvp);

  dim3 g2(S_ / 128, B_ * H_, 3);
  qkv_kernel<<<g2, 128, 0, stream>>>(xh, wqp, wkp, wvp, bq, bk, bv,
                                     qbuf, kbuf, vpk);

  dim3 g3(S_ / 128, B_ * H_);
  attn_kernel<<<g3, 128, 0, stream>>>(qbuf, kbuf, vpk, cbuf);

  dim3 g4(E_ / 64, (B_ * S_) / 128);
  oproj_kernel<<<g4, 128, 0, stream>>>(cbuf, woh, bo, out);
}
